// sparseSpatialAttention_52407190945934
// MI455X (gfx1250) — matrix operand order repaired and checked by reading, not compiled, not run on hardware
//
#include <hip/hip_runtime.h>
#include <hip/hip_bf16.h>

// MI455X / gfx1250, wave32. WMMA f16 -> f32 path throughout.
// Problem: B=16, T=12, N=512, D=128, scale = 1/sqrt(16) = 0.25, LN eps 1e-5.
//
// Structure: 6144 workgroups (192 bt-slices x 32 row-blocks of 16), 256 thr (8 waves).
// Each wave owns a 16x16 WMMA output tile; the shared 16x128 A tile is staged in
// LDS once per block; B fragments are preloaded into registers so WMMAs overlap
// with the global loads instead of serializing on s_wait_loadcnt 0.

typedef __attribute__((ext_vector_type(16))) _Float16 v16h;
typedef __attribute__((ext_vector_type(8)))  _Float16 v8h;
typedef __attribute__((ext_vector_type(8)))  float    v8f;

#define BT   192
#define NN   512
#define DD   128

static __device__ __forceinline__ v8f wmma32(v16h a, v16h b, v8f c) {
    // D = A(16x32 f16) * B(32x16 f16) + C(16x16 f32)
    return __builtin_amdgcn_wmma_f32_16x16x32_f16(false, a, false, b, (short)0, c, false, false);
}

// A fragment: rowbase points at this lane's row (row = lane&15).
// elements 0..7  : K = kbase + half*8 + e
// elements 8..15 : K = kbase + 16 + half*8 + (e-8)
static __device__ __forceinline__ v16h load_frag_a(const _Float16* rowbase, int kbase, int hf) {
    union { v16h v; v8h h[2]; } u;
    u.h[0] = *(const v8h*)(rowbase + kbase + hf * 8);
    u.h[1] = *(const v8h*)(rowbase + kbase + 16 + hf * 8);
    return u.v;
}

// B fragment from a row-major matrix M where B(KK,col) = M[row=col][kbase+KK]:
// elements 0..7 : K = half*8 + e ; elements 8..15 : K = 16 + half*8 + (e-8)
static __device__ __forceinline__ v16h load_frag_b(const _Float16* base, int row, int stride,
                                                   int kbase, int hf) {
    union { v16h v; v8h h[2]; } u;
    const _Float16* p = base + (size_t)row * stride + kbase + hf * 8;
    u.h[0] = *(const v8h*)(p);
    u.h[1] = *(const v8h*)(p + 16);
    return u.v;
}

// ---------------------------------------------------------------- prep
__global__ void prep_elems_kernel(const float* __restrict__ x, const float* __restrict__ adj,
                                  float* __restrict__ xr, _Float16* __restrict__ xh, int n) {
    int i = blockIdx.x * blockDim.x + threadIdx.x;
    if (i < n) {
        float v = x[i] + adj[i];
        xr[i] = v;
        xh[i] = (_Float16)v;
    }
}

// 98304 threads: transpose+convert six 128x128 weights, and zero colarg (BT*NN == 98304).
__global__ void prep_weights_kernel(const float* __restrict__ Wq, const float* __restrict__ Wk,
                                    const float* __restrict__ Wv, const float* __restrict__ Wo,
                                    const float* __restrict__ W1, const float* __restrict__ W2,
                                    _Float16* __restrict__ wt, unsigned long long* __restrict__ colarg) {
    int i = blockIdx.x * blockDim.x + threadIdx.x;
    colarg[i] = 0ull;
    int w = i >> 14;          // which weight
    int idx = i & 16383;
    int k = idx >> 7;         // fan_in index
    int n = idx & 127;        // fan_out index
    const float* Ws[6] = {Wq, Wk, Wv, Wo, W1, W2};
    wt[(w << 14) + n * 128 + k] = (_Float16)Ws[w][k * 128 + n];
}

// ---------------------------------------------------------------- QKV
__global__ __launch_bounds__(256) void qkv_kernel(const _Float16* __restrict__ xh,
                                                  const _Float16* __restrict__ wt,
                                                  const float* __restrict__ bq,
                                                  const float* __restrict__ bk,
                                                  const float* __restrict__ bv,
                                                  _Float16* __restrict__ Qh,
                                                  _Float16* __restrict__ Kh,
                                                  _Float16* __restrict__ Vth) {
    __shared__ __align__(16) _Float16 As[16 * 128];
    const int bt = blockIdx.x >> 5;
    const int m0 = (blockIdx.x & 31) << 4;
    const int wv = threadIdx.x >> 5, lane = threadIdx.x & 31;
    const int hf = lane >> 4, l15 = lane & 15;
    const int n0 = wv << 4;

    // stage contiguous 16x128 A tile once per block (one b128 per thread)
    ((v8h*)As)[threadIdx.x] =
        ((const v8h*)(xh + (size_t)(bt * NN + m0) * DD))[threadIdx.x];

    // preload all 12 B fragments (3 weights x 4 k-steps) to overlap with WMMA
    v16h b[12];
#pragma unroll
    for (int o = 0; o < 3; ++o)
#pragma unroll
        for (int k = 0; k < 4; ++k)
            b[o * 4 + k] = load_frag_b(wt + o * 16384, n0 + l15, 128, k * 32, hf);

    __syncthreads();
    const _Float16* arow = As + l15 * 128;
    v16h a[4];
#pragma unroll
    for (int k = 0; k < 4; ++k) a[k] = load_frag_a(arow, k * 32, hf);

    const float* bs[3] = {bq, bk, bv};
#pragma unroll
    for (int o = 0; o < 3; ++o) {
        v8f c = {};
#pragma unroll
        for (int k = 0; k < 4; ++k) c = wmma32(a[k], b[o * 4 + k], c);
        float bias = bs[o][n0 + l15];
#pragma unroll
        for (int r = 0; r < 8; ++r) {
            int row = m0 + r + 8 * hf;
            float y = c[r] + bias;
            if (o == 0)      Qh[((size_t)(bt * NN + row)) * DD + n0 + l15] = (_Float16)y;
            else if (o == 1) Kh[((size_t)(bt * NN + row)) * DD + n0 + l15] = (_Float16)y;
            else             Vth[(size_t)bt * (NN * DD) + (size_t)(n0 + l15) * NN + row] = (_Float16)y;
        }
    }
}

// ---------------------------------------------------------------- softmax stats (pass 1)
__global__ __launch_bounds__(256) void stats_kernel(const _Float16* __restrict__ Qh,
                                                    const _Float16* __restrict__ Kh,
                                                    float* __restrict__ rowlog) {
    __shared__ __align__(16) _Float16 Qs[16 * 128];
    __shared__ float pM[8][16];
    __shared__ float pS[8][16];
    const int bt = blockIdx.x >> 5;
    const int m0 = (blockIdx.x & 31) << 4;
    const int wv = threadIdx.x >> 5, lane = threadIdx.x & 31;
    const int hf = lane >> 4, l15 = lane & 15;

    ((v8h*)Qs)[threadIdx.x] =
        ((const v8h*)(Qh + (size_t)(bt * NN + m0) * DD))[threadIdx.x];
    __syncthreads();

    const _Float16* arow = Qs + l15 * 128;
    v16h a[4];
#pragma unroll
    for (int k = 0; k < 4; ++k) a[k] = load_frag_a(arow, k * 32, hf);

    const _Float16* kb = Kh + (size_t)bt * (NN * DD);
    float M[8], S[8];
#pragma unroll
    for (int r = 0; r < 8; ++r) { M[r] = -3.0e38f; S[r] = 0.0f; }

    for (int it = 0; it < 4; ++it) {
        int mc = (it * 8 + wv) << 4;
        v16h bf[4];
#pragma unroll
        for (int k = 0; k < 4; ++k) bf[k] = load_frag_b(kb, mc + l15, 128, k * 32, hf);
        v8f s = {};
#pragma unroll
        for (int k = 0; k < 4; ++k) s = wmma32(a[k], bf[k], s);
#pragma unroll
        for (int r = 0; r < 8; ++r) {
            float z = s[r] * 0.25f;
            float tm = z;
            tm = fmaxf(tm, __shfl_xor(tm, 1, 32));
            tm = fmaxf(tm, __shfl_xor(tm, 2, 32));
            tm = fmaxf(tm, __shfl_xor(tm, 4, 32));
            tm = fmaxf(tm, __shfl_xor(tm, 8, 32));
            float p = __expf(z - tm);
            float ts = p;
            ts += __shfl_xor(ts, 1, 32);
            ts += __shfl_xor(ts, 2, 32);
            ts += __shfl_xor(ts, 4, 32);
            ts += __shfl_xor(ts, 8, 32);
            float nm = fmaxf(M[r], tm);
            S[r] = S[r] * __expf(M[r] - nm) + ts * __expf(tm - nm);
            M[r] = nm;
        }
    }
#pragma unroll
    for (int r = 0; r < 8; ++r)
        if (l15 == r) { pM[wv][r + 8 * hf] = M[r]; pS[wv][r + 8 * hf] = S[r]; }
    __syncthreads();
    if (threadIdx.x < 16) {
        float Mt = -3.0e38f, St = 0.0f;
#pragma unroll
        for (int w = 0; w < 8; ++w) {
            float m2 = pM[w][threadIdx.x], s2 = pS[w][threadIdx.x];
            float nm = fmaxf(Mt, m2);
            St = St * __expf(Mt - nm) + s2 * __expf(m2 - nm);
            Mt = nm;
        }
        rowlog[bt * NN + m0 + threadIdx.x] = Mt + __logf(St);
    }
}

// ---------------------------------------------------------------- attention (pass 2) + col-argmax
__global__ __launch_bounds__(256) void attn_kernel(const _Float16* __restrict__ Qh,
                                                   const _Float16* __restrict__ Kh,
                                                   const _Float16* __restrict__ Vth,
                                                   const float* __restrict__ rowlog,
                                                   unsigned long long* __restrict__ colarg,
                                                   _Float16* __restrict__ atth) {
    __shared__ __align__(16) _Float16 Qs[16 * 128];
    __shared__ __align__(16) _Float16 P[16 * 128];
    __shared__ float rl[16];
    const int bt = blockIdx.x >> 5;
    const int m0 = (blockIdx.x & 31) << 4;
    const int wv = threadIdx.x >> 5, lane = threadIdx.x & 31;
    const int hf = lane >> 4, l15 = lane & 15;
    const int d0 = wv << 4;

    ((v8h*)Qs)[threadIdx.x] =
        ((const v8h*)(Qh + (size_t)(bt * NN + m0) * DD))[threadIdx.x];
    if (threadIdx.x < 16) rl[threadIdx.x] = rowlog[bt * NN + m0 + threadIdx.x];
    __syncthreads();

    const _Float16* arow = Qs + l15 * 128;
    v16h a[4];
#pragma unroll
    for (int k = 0; k < 4; ++k) a[k] = load_frag_a(arow, k * 32, hf);

    const _Float16* kb = Kh + (size_t)bt * (NN * DD);
    const _Float16* vb = Vth + (size_t)bt * (NN * DD);
    v8f acc = {};

    for (int it = 0; it < 4; ++it) {
        const int mc = it * 128 + (wv << 4);   // this wave's 16 score columns
        v16h bf[4];
#pragma unroll
        for (int k = 0; k < 4; ++k) bf[k] = load_frag_b(kb, mc + l15, 128, k * 32, hf);
        v8f s = {};
#pragma unroll
        for (int k = 0; k < 4; ++k) s = wmma32(a[k], bf[k], s);

        float bz = -3.0e38f;
        int bi = 0;
#pragma unroll
        for (int r = 0; r < 8; ++r) {
            float z = s[r] * 0.25f - rl[r + 8 * hf];
            float p = __expf(z);
            P[(r + 8 * hf) * 128 + (wv << 4) + l15] = (_Float16)p;
            if (z > bz) { bz = z; bi = m0 + r + 8 * hf; }
        }
        // merge the two halves (same column, different 8 rows)
        float oz = __shfl_xor(bz, 16, 32);
        int   oi = __shfl_xor(bi, 16, 32);
        if (oz > bz || (oz == bz && oi < bi)) { bz = oz; bi = oi; }
        if (hf == 0) {
            unsigned u = __float_as_uint(bz);
            unsigned key = (u & 0x80000000u) ? ~u : (u | 0x80000000u);
            unsigned long long pk = ((unsigned long long)key << 32) | (unsigned)(511 - bi);
            atomicMax(&colarg[bt * NN + mc + l15], pk);
        }
        __syncthreads();
        // attended += P(16x128) * V(128 x 16-col-tile); V B-fragments preloaded
        v16h vf[4];
#pragma unroll
        for (int kk = 0; kk < 4; ++kk)
            vf[kk] = load_frag_b(vb, d0 + l15, NN, it * 128 + kk * 32, hf);
        const _Float16* prow = P + l15 * 128;
#pragma unroll
        for (int kk = 0; kk < 4; ++kk) {
            v16h pa = load_frag_a(prow, kk * 32, hf);
            acc = wmma32(pa, vf[kk], acc);
        }
        __syncthreads();
    }
#pragma unroll
    for (int r = 0; r < 8; ++r)
        atth[(size_t)(bt * NN + m0 + r + 8 * hf) * DD + d0 + l15] = (_Float16)acc[r];
}

// ---------------------------------------------------------------- gather + Wo + residual + LN
__global__ __launch_bounds__(256) void gwo_ln_kernel(const _Float16* __restrict__ atth,
                                                     const unsigned long long* __restrict__ colarg,
                                                     const _Float16* __restrict__ wto,
                                                     const float* __restrict__ bo,
                                                     const float* __restrict__ xr,
                                                     _Float16* __restrict__ vh) {
    __shared__ __align__(16) _Float16 As[16 * 128];
    __shared__ float pSum[8][16];
    __shared__ float pSq[8][16];
    __shared__ float mu[16];
    __shared__ float rs[16];
    const int bt = blockIdx.x >> 5;
    const int m0 = (blockIdx.x & 31) << 4;
    const int wv = threadIdx.x >> 5, lane = threadIdx.x & 31;
    const int hf = lane >> 4, l15 = lane & 15;
    const int n0 = wv << 4;

    // hard gather while staging: row r of the A tile comes from attended[cp[m0+r]]
    {
        int row = threadIdx.x >> 4;       // 0..15
        int seg = threadIdx.x & 15;       // 16B segment within the row
        int srow = 511 - (int)(colarg[bt * NN + m0 + row] & 0xffffffffu);
        ((v8h*)As)[threadIdx.x] =
            *(const v8h*)(atth + (size_t)(bt * NN + srow) * DD + seg * 8);
    }
    v16h b[4];
#pragma unroll
    for (int k = 0; k < 4; ++k) b[k] = load_frag_b(wto, n0 + l15, 128, k * 32, hf);
    __syncthreads();

    const _Float16* arow = As + l15 * 128;
    v16h a[4];
#pragma unroll
    for (int k = 0; k < 4; ++k) a[k] = load_frag_a(arow, k * 32, hf);
    v8f c = {};
#pragma unroll
    for (int k = 0; k < 4; ++k) c = wmma32(a[k], b[k], c);

    float bias = bo[n0 + l15];
    float y[8];
#pragma unroll
    for (int r = 0; r < 8; ++r)
        y[r] = c[r] + bias + xr[(size_t)(bt * NN + m0 + r + 8 * hf) * DD + n0 + l15];

#pragma unroll
    for (int r = 0; r < 8; ++r) {
        float s = y[r], q = y[r] * y[r];
        s += __shfl_xor(s, 1, 32);  q += __shfl_xor(q, 1, 32);
        s += __shfl_xor(s, 2, 32);  q += __shfl_xor(q, 2, 32);
        s += __shfl_xor(s, 4, 32);  q += __shfl_xor(q, 4, 32);
        s += __shfl_xor(s, 8, 32);  q += __shfl_xor(q, 8, 32);
        if (l15 == r) { pSum[wv][r + 8 * hf] = s; pSq[wv][r + 8 * hf] = q; }
    }
    __syncthreads();
    if (threadIdx.x < 16) {
        float s = 0.0f, q = 0.0f;
#pragma unroll
        for (int w = 0; w < 8; ++w) { s += pSum[w][threadIdx.x]; q += pSq[w][threadIdx.x]; }
        float m = s * (1.0f / 128.0f);
        float v = q * (1.0f / 128.0f) - m * m;
        mu[threadIdx.x] = m;
        rs[threadIdx.x] = rsqrtf(v + 1e-5f);
    }
    __syncthreads();
#pragma unroll
    for (int r = 0; r < 8; ++r) {
        int lr = r + 8 * hf;
        float v = (y[r] - mu[lr]) * rs[lr];
        vh[(size_t)(bt * NN + m0 + lr) * DD + n0 + l15] = (_Float16)v;
    }
}

// ---------------------------------------------------------------- FFN
__global__ __launch_bounds__(256) void ffn1_kernel(const _Float16* __restrict__ vh,
                                                   const _Float16* __restrict__ wt1,
                                                   const float* __restrict__ b1,
                                                   _Float16* __restrict__ hh) {
    __shared__ __align__(16) _Float16 As[16 * 128];
    const int bt = blockIdx.x >> 5;
    const int m0 = (blockIdx.x & 31) << 4;
    const int wv = threadIdx.x >> 5, lane = threadIdx.x & 31;
    const int hf = lane >> 4, l15 = lane & 15;
    const int n0 = wv << 4;

    ((v8h*)As)[threadIdx.x] =
        ((const v8h*)(vh + (size_t)(bt * NN + m0) * DD))[threadIdx.x];
    v16h b[4];
#pragma unroll
    for (int k = 0; k < 4; ++k) b[k] = load_frag_b(wt1, n0 + l15, 128, k * 32, hf);
    __syncthreads();

    const _Float16* arow = As + l15 * 128;
    v16h a[4];
#pragma unroll
    for (int k = 0; k < 4; ++k) a[k] = load_frag_a(arow, k * 32, hf);
    v8f c = {};
#pragma unroll
    for (int k = 0; k < 4; ++k) c = wmma32(a[k], b[k], c);

    float bias = b1[n0 + l15];
#pragma unroll
    for (int r = 0; r < 8; ++r) {
        float y = fmaxf(c[r] + bias, 0.0f);   // ReLU
        hh[(size_t)(bt * NN + m0 + r + 8 * hf) * DD + n0 + l15] = (_Float16)y;
    }
}

__global__ __launch_bounds__(256) void ffn2_kernel(const _Float16* __restrict__ hh,
                                                   const _Float16* __restrict__ wt2,
                                                   const float* __restrict__ b2,
                                                   float* __restrict__ out) {
    __shared__ __align__(16) _Float16 As[16 * 128];
    const int bt = blockIdx.x >> 5;
    const int m0 = (blockIdx.x & 31) << 4;
    const int wv = threadIdx.x >> 5, lane = threadIdx.x & 31;
    const int hf = lane >> 4, l15 = lane & 15;
    const int n0 = wv << 4;

    ((v8h*)As)[threadIdx.x] =
        ((const v8h*)(hh + (size_t)(bt * NN + m0) * DD))[threadIdx.x];
    v16h b[4];
#pragma unroll
    for (int k = 0; k < 4; ++k) b[k] = load_frag_b(wt2, n0 + l15, 128, k * 32, hf);
    __syncthreads();

    const _Float16* arow = As + l15 * 128;
    v16h a[4];
#pragma unroll
    for (int k = 0; k < 4; ++k) a[k] = load_frag_a(arow, k * 32, hf);
    v8f c = {};
#pragma unroll
    for (int k = 0; k < 4; ++k) c = wmma32(a[k], b[k], c);

    float bias = b2[n0 + l15];
#pragma unroll
    for (int r = 0; r < 8; ++r)
        out[(size_t)(bt * NN + m0 + r + 8 * hf) * DD + n0 + l15] = c[r] + bias;
}

// ---------------------------------------------------------------- launch
extern "C" void kernel_launch(void* const* d_in, const int* in_sizes, int n_in,
                              void* d_out, int out_size, void* d_ws, size_t ws_size,
                              hipStream_t stream) {
    const float* x   = (const float*)d_in[0];
    const float* adj = (const float*)d_in[1];
    const float* Wq  = (const float*)d_in[2];
    const float* bq  = (const float*)d_in[3];
    const float* Wk  = (const float*)d_in[4];
    const float* bk  = (const float*)d_in[5];
    const float* Wv  = (const float*)d_in[6];
    const float* bv  = (const float*)d_in[7];
    const float* Wo  = (const float*)d_in[8];
    const float* bo  = (const float*)d_in[9];
    const float* W1  = (const float*)d_in[10];
    const float* b1  = (const float*)d_in[11];
    const float* W2  = (const float*)d_in[12];
    const float* b2  = (const float*)d_in[13];
    float* out = (float*)d_out;

    const size_t elems = (size_t)BT * NN * DD;          // 12,582,912
    char* p = (char*)d_ws;
    float*              xr     = (float*)p;                 p += elems * sizeof(float);
    _Float16*           xh     = (_Float16*)p;              p += elems * sizeof(_Float16);
    _Float16*           Qh     = (_Float16*)p;              p += elems * sizeof(_Float16);
    _Float16*           Kh     = (_Float16*)p;              p += elems * sizeof(_Float16);
    _Float16*           Vth    = (_Float16*)p;              p += elems * sizeof(_Float16);
    _Float16*           atth   = (_Float16*)p;              p += elems * sizeof(_Float16);
    _Float16*           vh     = (_Float16*)p;              p += elems * sizeof(_Float16);
    _Float16*           hh     = (_Float16*)p;              p += elems * sizeof(_Float16);
    float*              rowlog = (float*)p;                 p += (size_t)BT * NN * sizeof(float);
    unsigned long long* colarg = (unsigned long long*)p;    p += (size_t)BT * NN * sizeof(unsigned long long);
    _Float16*           wt     = (_Float16*)p;              p += 6 * 128 * 128 * sizeof(_Float16);

    const int nelem = (int)elems;
    prep_elems_kernel<<<(nelem + 255) / 256, 256, 0, stream>>>(x, adj, xr, xh, nelem);
    prep_weights_kernel<<<(BT * NN) / 256, 256, 0, stream>>>(Wq, Wk, Wv, Wo, W1, W2, wt, colarg);

    const int blocks = BT * (NN / 16);                  // 6144
    qkv_kernel  <<<blocks, 256, 0, stream>>>(xh, wt, bq, bk, bv, Qh, Kh, Vth);
    stats_kernel<<<blocks, 256, 0, stream>>>(Qh, Kh, rowlog);
    attn_kernel <<<blocks, 256, 0, stream>>>(Qh, Kh, Vth, rowlog, colarg, atth);
    gwo_ln_kernel<<<blocks, 256, 0, stream>>>(atth, colarg, wt + 3 * 16384, bo, xr, vh);
    ffn1_kernel <<<blocks, 256, 0, stream>>>(vh, wt + 4 * 16384, b1, hh);
    ffn2_kernel <<<blocks, 256, 0, stream>>>(hh, wt + 5 * 16384, b2, out);
}
